// Encoder_10067403341857
// MI455X (gfx1250) — matrix operand order repaired and checked by reading, not compiled, not run on hardware
//
#include <hip/hip_runtime.h>

// ---------------------------------------------------------------------------
// CDNA5 (gfx1250) transformer encoder forward.  wave32, WMMA bf16 -> f32.
// ---------------------------------------------------------------------------

typedef __bf16 bf16_t;
typedef __attribute__((ext_vector_type(2)))  bf16_t bf16x2;
typedef __attribute__((ext_vector_type(16))) bf16_t v16bf;
typedef __attribute__((ext_vector_type(8)))  float  v8f;
typedef __attribute__((ext_vector_type(4)))  float  f32x4;
typedef __attribute__((ext_vector_type(4)))  unsigned int u32x4;
typedef __attribute__((ext_vector_type(4)))  int v4i_t;

constexpr int BB = 8;        // batch
constexpr int SS = 512;      // seq len
constexpr int DD = 1024;     // hidden
constexpr int HH = 16;       // heads
constexpr int HD = 64;       // head dim
constexpr int FF = 4096;     // ffn dim
constexpr int LL = 6;        // layers
constexpr int MR = BB * SS;  // 4096 rows

#define DEV __device__ __forceinline__

#if __has_builtin(__builtin_amdgcn_global_load_async_to_lds_b128) && \
    __has_builtin(__builtin_amdgcn_s_wait_asynccnt)
#define USE_ASYNC_LDS 1
typedef __attribute__((address_space(1))) v4i_t* as1_v4i_p;
typedef __attribute__((address_space(3))) v4i_t* as3_v4i_p;
#else
#define USE_ASYNC_LDS 0
#endif

// 16-byte global->LDS copy; async (ASYNCcnt) when the toolchain exposes it.
DEV void copy_b128_to_lds(const void* gp, void* lp) {
#if USE_ASYNC_LDS
    __builtin_amdgcn_global_load_async_to_lds_b128(
        (as1_v4i_p)(void*)gp, (as3_v4i_p)lp, 0, 0);
#else
    *(u32x4*)lp = *(const u32x4*)gp;
#endif
}

DEV void async_copies_wait() {
#if USE_ASYNC_LDS
    __builtin_amdgcn_s_wait_asynccnt(0);   // s_wait_asynccnt 0
#endif
}

DEV v8f wmma_bf16(v16bf a, v16bf b, v8f c) {
    // v_wmma_f32_16x16x32_bf16  (neg_a, A, neg_b, B, c_mod, C, reuse_a, reuse_b)
    return __builtin_amdgcn_wmma_f32_16x16x32_bf16(false, a, false, b,
                                                   (short)0, c, false, false);
}

// A-fragment: 16x32 bf16, source row-major [16][ld].  ISA 7.12.2:
// lanes 0-15 row=lane, V0-3: K=0..7, V4-7: K=16..23 ; lanes 16-31: +8.
DEV v16bf load_a_frag(const bf16_t* __restrict__ base, int ld, int lane) {
    const int r  = lane & 15;
    const int hi = (lane >> 4) & 1;
    const bf16_t* p = base + (size_t)r * ld;
    v16bf a;
#pragma unroll
    for (int v = 0; v < 8; ++v) {
        int k = ((v < 4) ? (2 * v) : (16 + 2 * (v - 4))) + (hi ? 8 : 0);
        bf16x2 t = *(const bf16x2*)(p + k);
        a[2 * v]     = t.x;
        a[2 * v + 1] = t.y;
    }
    return a;
}

// B-fragment: 32x16 bf16 from TRANSPOSED storage [N][K] stride ld.
// lanes 0-15: V0-3: K=0..7, V4-7: K=16..23 ; lanes 16-31: +8 (matches A).
DEV v16bf load_b_frag(const bf16_t* __restrict__ base, int ld, int lane) {
    const int n  = lane & 15;
    const int hi = (lane >> 4) & 1;
    const bf16_t* p = base + (size_t)n * ld;
    v16bf b;
#pragma unroll
    for (int v = 0; v < 8; ++v) {
        int k = ((v < 4) ? (2 * v) : (16 + 2 * (v - 4))) + (hi ? 8 : 0);
        bf16x2 t = *(const bf16x2*)(p + k);
        b[2 * v]     = t.x;
        b[2 * v + 1] = t.y;
    }
    return b;
}

// ---------------------------------------------------------------------------
// Embedding:  x = tok_emb[src]*sqrt(D) + pos_emb[s]
// ---------------------------------------------------------------------------
__global__ __launch_bounds__(256) void embed_kernel(
        const int* __restrict__ src, const float* __restrict__ tok,
        const float* __restrict__ pos, float* __restrict__ xf,
        bf16_t* __restrict__ xb) {
    int row = blockIdx.x;                 // b*S + s
    int s   = row & (SS - 1);
    int t   = src[row];
    const float* te = tok + (size_t)t * DD;
    const float* pe = pos + (size_t)s * DD;
#pragma unroll
    for (int i = 0; i < DD / 256; ++i) {
        int c = threadIdx.x + i * 256;
        float v = te[c] * 32.0f + pe[c];  // sqrt(1024) = 32
        xf[(size_t)row * DD + c] = v;
        xb[(size_t)row * DD + c] = (bf16_t)v;
    }
}

// ---------------------------------------------------------------------------
// LayerNorm (row = 1024), writes fp32 residual stream + bf16 matmul input
// ---------------------------------------------------------------------------
__global__ __launch_bounds__(256) void layernorm_kernel(
        const float* __restrict__ in, const float* __restrict__ g,
        const float* __restrict__ be, float* __restrict__ xf,
        bf16_t* __restrict__ xb) {
    __shared__ float ws1[8], ws2[8];
    int row = blockIdx.x;
    int lane = threadIdx.x & 31, wave = threadIdx.x >> 5;
    const float* p = in + (size_t)row * DD;
    float vals[DD / 256];
    float s = 0.f, s2 = 0.f;
#pragma unroll
    for (int i = 0; i < DD / 256; ++i) {
        float v = p[threadIdx.x + i * 256];
        vals[i] = v; s += v; s2 += v * v;
    }
#pragma unroll
    for (int m = 16; m >= 1; m >>= 1) {
        s  += __shfl_xor(s,  m, 32);
        s2 += __shfl_xor(s2, m, 32);
    }
    if (lane == 0) { ws1[wave] = s; ws2[wave] = s2; }
    __syncthreads();
    s = 0.f; s2 = 0.f;
#pragma unroll
    for (int w = 0; w < 8; ++w) { s += ws1[w]; s2 += ws2[w]; }
    float mean = s * (1.0f / DD);
    float var  = s2 * (1.0f / DD) - mean * mean;
    float rinv = rsqrtf(var + 1e-5f);
#pragma unroll
    for (int i = 0; i < DD / 256; ++i) {
        int c = threadIdx.x + i * 256;
        float o = (vals[i] - mean) * rinv * g[c] + be[c];
        xf[(size_t)row * DD + c] = o;
        xb[(size_t)row * DD + c] = (bf16_t)o;
    }
}

// ---------------------------------------------------------------------------
// Generic WMMA GEMM:  C[M,N] = epilogue(A_bf16[M,K] * W_f32[K,N])
// block tile 128x128, K-chunk 32, 256 threads = 8 waves.
// Each wave owns 2 M-tiles x 4 N-tiles -> 8 WMMA per K-chunk, B-frags reused.
// A tile staged via GLOBAL_LOAD_ASYNC_TO_LDS_B128 when available.
// ---------------------------------------------------------------------------
template <bool BIAS, bool RES, bool RELU, bool OUTF, bool OUTB>
__global__ __launch_bounds__(256) void gemm_wmma(
        const bf16_t* __restrict__ A, const float* __restrict__ W,
        const float* __restrict__ bias, const float* __restrict__ res,
        float* __restrict__ Cf, bf16_t* __restrict__ Cb,
        int M, int N, int K) {
    constexpr int BM = 128, BN = 128, BK = 32;
    constexpr int LDA = BK + 8;   // 40 bf16 -> 80B rows, keeps b128 alignment
    constexpr int LDW = BK + 8;
    __shared__ bf16_t As[BM * LDA];   // 10 KB
    __shared__ bf16_t Wt[BN * LDW];   // 10 KB, stored transposed [n][k]

    const int tid  = threadIdx.x;
    const int lane = tid & 31, wave = tid >> 5;
    const int hi   = (lane >> 4) & 1;
    const int n0   = blockIdx.x * BN;
    const int m0   = blockIdx.y * BM;
    const int mt0  = (wave >> 1) * 2;  // 0,2,4,6  (2 M sub-tiles per wave)
    const int ntb  = (wave & 1) * 4;   // 0 / 4    (4 N sub-tiles per wave)

    v8f acc[2][4] = {};

    for (int k0 = 0; k0 < K; k0 += BK) {
        // ---- stage A tile 128x32 bf16: 2 async b128 per thread ----------
#pragma unroll
        for (int i = 0; i < 2; ++i) {
            int idx = tid * 2 + i;        // 0..511 b128 units
            int row = idx >> 2, ch = idx & 3;
            copy_b128_to_lds(A + (size_t)(m0 + row) * K + k0 + ch * 8,
                             &As[row * LDA + ch * 8]);
        }
        // ---- stage W tile 32x128 f32 -> bf16 transposed [n][k] ----------
#pragma unroll
        for (int i = 0; i < 4; ++i) {
            int idx = tid + i * 256;          // 1024 float4 units
            int kk  = idx >> 5;               // 0..31
            int nn  = (idx & 31) * 4;         // 0..124
            f32x4 w = *(const f32x4*)(W + (size_t)(k0 + kk) * N + n0 + nn);
            Wt[(nn + 0) * LDW + kk] = (bf16_t)w.x;
            Wt[(nn + 1) * LDW + kk] = (bf16_t)w.y;
            Wt[(nn + 2) * LDW + kk] = (bf16_t)w.z;
            Wt[(nn + 3) * LDW + kk] = (bf16_t)w.w;
        }
        // prefetch next K-chunk of W into L2 (global_prefetch)
        if (k0 + BK < K)
            __builtin_prefetch(W + (size_t)(k0 + BK + (tid >> 3)) * N + n0, 0, 1);
        async_copies_wait();
        __syncthreads();

        v16bf af0 = load_a_frag(&As[(mt0 * 16) * LDA], LDA, lane);
        v16bf af1 = load_a_frag(&As[((mt0 + 1) * 16) * LDA], LDA, lane);
#pragma unroll
        for (int j = 0; j < 4; ++j) {
            v16bf bfr = load_b_frag(&Wt[((ntb + j) * 16) * LDW], LDW, lane);
            acc[0][j] = wmma_bf16(af0, bfr, acc[0][j]);
            acc[1][j] = wmma_bf16(af1, bfr, acc[1][j]);
        }
        __syncthreads();
    }

    // ---- epilogue: C layout = row (r + 8*hi), col (lane&15) -------------
    const int colw = lane & 15;
#pragma unroll
    for (int j = 0; j < 4; ++j) {
        int col = n0 + (ntb + j) * 16 + colw;
        float bb = BIAS ? bias[col] : 0.f;
#pragma unroll
        for (int i = 0; i < 2; ++i)
#pragma unroll
            for (int r = 0; r < 8; ++r) {
                int row = m0 + (mt0 + i) * 16 + r + (hi ? 8 : 0);
                float v = acc[i][j][r] + bb;
                if (RES)  v += res[(size_t)row * N + col];
                if (RELU) v = v > 0.f ? v : 0.f;
                if (OUTF) Cf[(size_t)row * N + col] = v;
                if (OUTB) Cb[(size_t)row * N + col] = (bf16_t)v;
            }
    }
}

// ---------------------------------------------------------------------------
// Fused flash attention: per block = (b, h, 128 q rows); per wave = 16 q rows
// Online softmax, Q frags resident in VGPRs, V transposed once into LDS.
// ---------------------------------------------------------------------------
__global__ __launch_bounds__(256) void attn_kernel(
        const bf16_t* __restrict__ q, const bf16_t* __restrict__ k,
        const bf16_t* __restrict__ v, const int* __restrict__ mask,
        bf16_t* __restrict__ ctx) {
    constexpr int LVS = SS + 8;               // 520
    __shared__ bf16_t Vt[HD * LVS];           // 65 KB  V^T [d][s]
    __shared__ bf16_t Pst[8 * 16 * 32];       // 8 KB per-wave P staging

    const int blk  = blockIdx.x;              // b*64 + h*4 + qblk
    const int qblk = blk & 3;
    const int h    = (blk >> 2) & 15;
    const int b    = blk >> 6;
    const int lane = threadIdx.x & 31, wave = threadIdx.x >> 5;
    const int hi   = (lane >> 4) & 1;
    const size_t baseBH = (size_t)b * SS * DD + (size_t)h * HD;

    // ---- stage V^T into LDS --------------------------------------------
    for (int s0 = threadIdx.x; s0 < SS; s0 += 256) {
        const bf16_t* vp = v + baseBH + (size_t)s0 * DD;
#pragma unroll
        for (int d = 0; d < HD; ++d) Vt[d * LVS + s0] = vp[d];
    }
    __syncthreads();

    // ---- resident Q fragments (16 rows x 64 d) -------------------------
    const int qrow0 = qblk * 128 + wave * 16;
    const bf16_t* qbase = q + baseBH + (size_t)qrow0 * DD;
    v16bf qf0 = load_a_frag(qbase,      DD, lane);   // d 0..31
    v16bf qf1 = load_a_frag(qbase + 32, DD, lane);   // d 32..63

    float mrow[8], lrow[8];
    v8f o[4] = {};
#pragma unroll
    for (int r = 0; r < 8; ++r) { mrow[r] = -3.0e38f; lrow[r] = 0.f; }

    bf16_t* pst = &Pst[wave * 16 * 32];

    for (int c = 0; c < SS / 32; ++c) {       // 16 chunks of 32 kv rows
        const int kcol0 = c * 32;
        v8f st[2];
#pragma unroll
        for (int t2 = 0; t2 < 2; ++t2) {      // S = Q * K^T  (2 wmma each)
            const bf16_t* kb = k + baseBH + (size_t)(kcol0 + t2 * 16) * DD;
            v8f z = {};
            z = wmma_bf16(qf0, load_b_frag(kb,      DD, lane), z);
            z = wmma_bf16(qf1, load_b_frag(kb + 32, DD, lane), z);
            st[t2] = z;
        }
        // scale 1/sqrt(D) then mask (src_mask[b,0,0,kpos]==0 -> -1e10)
#pragma unroll
        for (int t2 = 0; t2 < 2; ++t2) {
            int kpos = kcol0 + t2 * 16 + (lane & 15);
            bool msk = (mask[b * SS + kpos] == 0);
#pragma unroll
            for (int r = 0; r < 8; ++r) {
                float e = st[t2][r] * 0.03125f;
                st[t2][r] = msk ? -1.0e10f : e;
            }
        }
        // online softmax: row max over 32 cols (16-lane half-wave reduce)
        float mnew[8], corr[8];
#pragma unroll
        for (int r = 0; r < 8; ++r) {
            float mx = fmaxf(st[0][r], st[1][r]);
            mx = fmaxf(mx, __shfl_xor(mx, 1, 32));
            mx = fmaxf(mx, __shfl_xor(mx, 2, 32));
            mx = fmaxf(mx, __shfl_xor(mx, 4, 32));
            mx = fmaxf(mx, __shfl_xor(mx, 8, 32));
            mnew[r] = fmaxf(mrow[r], mx);
            corr[r] = __expf(mrow[r] - mnew[r]);
            mrow[r] = mnew[r];
        }
#pragma unroll
        for (int t2 = 0; t2 < 2; ++t2)
#pragma unroll
            for (int r = 0; r < 8; ++r)
                st[t2][r] = __expf(st[t2][r] - mnew[r]);
#pragma unroll
        for (int r = 0; r < 8; ++r) {
            float ps = st[0][r] + st[1][r];
            ps += __shfl_xor(ps, 1, 32);
            ps += __shfl_xor(ps, 2, 32);
            ps += __shfl_xor(ps, 4, 32);
            ps += __shfl_xor(ps, 8, 32);
            lrow[r] = lrow[r] * corr[r] + ps;
        }
        // C-layout -> A-layout bounce through wave-local LDS (in-order DS)
#pragma unroll
        for (int t2 = 0; t2 < 2; ++t2)
#pragma unroll
            for (int r = 0; r < 8; ++r)
                pst[(r + (hi ? 8 : 0)) * 32 + t2 * 16 + (lane & 15)] =
                    (bf16_t)st[t2][r];
        // rescale O then accumulate P * V
#pragma unroll
        for (int j = 0; j < 4; ++j)
#pragma unroll
            for (int r = 0; r < 8; ++r) o[j][r] *= corr[r];
        v16bf pf = load_a_frag(pst, 32, lane);
#pragma unroll
        for (int j = 0; j < 4; ++j) {
            v16bf vf = load_b_frag(&Vt[(j * 16) * LVS + kcol0], LVS, lane);
            o[j] = wmma_bf16(pf, vf, o[j]);
        }
    }

    // ---- epilogue: ctx[b, qrow, h*64 + d] = O / l ----------------------
    bf16_t* cb = ctx + baseBH + (size_t)qrow0 * DD;
#pragma unroll
    for (int j = 0; j < 4; ++j)
#pragma unroll
        for (int r = 0; r < 8; ++r) {
            int row = r + (hi ? 8 : 0);
            float val = o[j][r] / lrow[r];
            cb[(size_t)row * DD + j * 16 + (lane & 15)] = (bf16_t)val;
        }
}

// ---------------------------------------------------------------------------
// Host orchestration
// ---------------------------------------------------------------------------
extern "C" void kernel_launch(void* const* d_in, const int* in_sizes, int n_in,
                              void* d_out, int out_size, void* d_ws,
                              size_t ws_size, hipStream_t stream) {
    const int*   src  = (const int*)d_in[0];
    const int*   msk  = (const int*)d_in[1];
    const float* tok  = (const float*)d_in[2];
    const float* pos  = (const float*)d_in[3];
    const float* Wq   = (const float*)d_in[4];
    const float* bq   = (const float*)d_in[5];
    const float* Wk   = (const float*)d_in[6];
    const float* bk   = (const float*)d_in[7];
    const float* Wv   = (const float*)d_in[8];
    const float* bv   = (const float*)d_in[9];
    const float* Wo   = (const float*)d_in[10];
    const float* bo   = (const float*)d_in[11];
    const float* g1   = (const float*)d_in[12];
    const float* be1  = (const float*)d_in[13];
    const float* g2   = (const float*)d_in[14];
    const float* be2  = (const float*)d_in[15];
    const float* W1   = (const float*)d_in[16];
    const float* bf1  = (const float*)d_in[17];
    const float* W2   = (const float*)d_in[18];
    const float* bf2  = (const float*)d_in[19];

    char* ws = (char*)d_ws;
    auto alloc = [&](size_t bytes) {
        char* p = ws;
        ws += (bytes + 255) & ~(size_t)255;
        return p;
    };
    float*  xf   = (float*)alloc((size_t)MR * DD * 4);
    float*  tf   = (float*)alloc((size_t)MR * DD * 4);
    bf16_t* xb   = (bf16_t*)alloc((size_t)MR * DD * 2);
    bf16_t* qb   = (bf16_t*)alloc((size_t)MR * DD * 2);
    bf16_t* kb2  = (bf16_t*)alloc((size_t)MR * DD * 2);
    bf16_t* vb2  = (bf16_t*)alloc((size_t)MR * DD * 2);
    bf16_t* cxb  = (bf16_t*)alloc((size_t)MR * DD * 2);
    bf16_t* hb   = (bf16_t*)alloc((size_t)MR * FF * 2);

    embed_kernel<<<MR, 256, 0, stream>>>(src, tok, pos, xf, xb);

    const dim3 gD(DD / 128, MR / 128);   // N=1024 GEMMs
    const dim3 gF(FF / 128, MR / 128);   // N=4096 GEMM (FFN up)

    for (int i = 0; i < LL; ++i) {
        const float* wq = Wq + (size_t)i * DD * DD;
        const float* wk = Wk + (size_t)i * DD * DD;
        const float* wv = Wv + (size_t)i * DD * DD;
        const float* wo = Wo + (size_t)i * DD * DD;
        const float* w1 = W1 + (size_t)i * DD * FF;
        const float* w2 = W2 + (size_t)i * FF * DD;

        // Q, K, V projections (bias, bf16 out)
        gemm_wmma<true, false, false, false, true><<<gD, 256, 0, stream>>>(
            xb, wq, bq + i * DD, nullptr, nullptr, qb, MR, DD, DD);
        gemm_wmma<true, false, false, false, true><<<gD, 256, 0, stream>>>(
            xb, wk, bk + i * DD, nullptr, nullptr, kb2, MR, DD, DD);
        gemm_wmma<true, false, false, false, true><<<gD, 256, 0, stream>>>(
            xb, wv, bv + i * DD, nullptr, nullptr, vb2, MR, DD, DD);

        // fused attention
        attn_kernel<<<BB * HH * (SS / 128), 256, 0, stream>>>(
            qb, kb2, vb2, msk, cxb);

        // output projection + residual, then LN1
        gemm_wmma<true, true, false, true, false><<<gD, 256, 0, stream>>>(
            cxb, wo, bo + i * DD, xf, tf, nullptr, MR, DD, DD);
        layernorm_kernel<<<MR, 256, 0, stream>>>(
            tf, g1 + i * DD, be1 + i * DD, xf, xb);

        // FFN: relu(x W1 + b1) W2 + b2 + residual, then LN2
        gemm_wmma<true, false, true, false, true><<<gF, 256, 0, stream>>>(
            xb, w1, bf1 + i * FF, nullptr, nullptr, hb, MR, FF, DD);
        gemm_wmma<true, true, false, true, false><<<gD, 256, 0, stream>>>(
            hb, w2, bf2 + i * DD, xf, tf, nullptr, MR, DD, FF);
        layernorm_kernel<<<MR, 256, 0, stream>>>(
            tf, g2 + i * DD, be2 + i * DD, xf, xb);
    }

    (void)hipMemcpyAsync(d_out, xf, (size_t)MR * DD * sizeof(float),
                         hipMemcpyDeviceToDevice, stream);
}
